// TransformerBlock_8074538516572
// MI455X (gfx1250) — compile-verified, matrix operand order checked by reading, not run on hardware
//
#include <hip/hip_runtime.h>
#include <hip/hip_bf16.h>
#include <math.h>

// ---------------- problem constants (from reference) ----------------
#define SQ   1024   // sequence length S
#define DM   1024   // model dim D
#define NHD  8      // heads
#define DH_  128
#define DR_  64
#define DCQ_ 768
#define DC_  256
#define INTER_ 512
#define NE   8      // experts
#define QKD  192    // DH + DR

typedef __attribute__((ext_vector_type(16))) _Float16 v16h;
typedef __attribute__((ext_vector_type(8)))  _Float16 v8h;
typedef __attribute__((ext_vector_type(8)))  float    v8f;
typedef __attribute__((ext_vector_type(4)))  float    v4fx;

#define SHUF16(lo, hi) __builtin_shufflevector(lo, hi, 0,1,2,3,4,5,6,7,8,9,10,11,12,13,14,15)

#define LDSPAD 40   // f16 row stride: 80B -> bank starts row*20 mod 64, conflict-free b128 reads

// =====================================================================
// WMMA GEMM, NT:  C[M,N] = A[M,K] @ W[N,K]^T  (+bias)(act)(+add)(*gate)(+=)
// Block = 256 threads = 8 waves. Block tile 128(M) x 64(N), K-step 32.
// A(128x32) and B(64x32) staged in LDS as f16 (converted once per block).
// Each wave computes a 16x64 strip: 4 WMMAs per K-step, A fragment reused 4x,
// B fragments shared by all 8 waves through LDS.
// grid = (N/64, M/128); K multiple of 32; M,N exact (true at all call sites).
// =====================================================================
__global__ void gemm_nt(const float* __restrict__ A, int lda,
                        const float* __restrict__ W, int ldw,
                        const float* __restrict__ bias,
                        float* __restrict__ C, int ldc,
                        int M, int N, int K,
                        int act,                       // 0=none, 1=silu
                        const float* __restrict__ addend, int ldadd,
                        const float* __restrict__ rowscale, int rsStride,
                        int accumulate)
{
    __shared__ _Float16 aS[128][LDSPAD];
    __shared__ _Float16 bS[64][LDSPAD];

    const int t    = threadIdx.x;
    const int lane = t & 31;
    const int wave = t >> 5;
    const int half = lane >> 4;
    const int r16  = lane & 15;
    const int m0b  = blockIdx.y * 128;
    const int n0b  = blockIdx.x * 64;

    // staging assignments (per thread, fixed across K loop)
    const int arow = t >> 1, ahalf = t & 1;          // A: 128 rows x 2 half-K chunks of 16
    const int brow = t >> 2, bq    = t & 3;          // B: 64 rows x 4 chunks of 8
    const float* agp = A + (size_t)(m0b + arow) * lda + ahalf * 16;
    const float* bgp = W + (size_t)(n0b + brow) * ldw + bq * 8;

    v8f acc[4] = {};

    for (int k0 = 0; k0 < K; k0 += 32) {
        __syncthreads();                 // previous iteration's fragment reads done
        // ---- stage A tile (16 floats -> f16) ----
        {
            const v4fx* p = (const v4fx*)(agp + k0);
            __builtin_prefetch(agp + k0 + 32, 0, 3);   // global_prefetch_b8
            v4fx f0 = p[0], f1 = p[1], f2 = p[2], f3 = p[3];
            _Float16* d = &aS[arow][ahalf * 16];
#pragma unroll
            for (int i = 0; i < 4; ++i) {
                d[i]      = (_Float16)f0[i];
                d[4 + i]  = (_Float16)f1[i];
                d[8 + i]  = (_Float16)f2[i];
                d[12 + i] = (_Float16)f3[i];
            }
        }
        // ---- stage B tile (8 floats -> f16) ----
        {
            const v4fx* p = (const v4fx*)(bgp + k0);
            __builtin_prefetch(bgp + k0 + 32, 0, 3);
            v4fx f0 = p[0], f1 = p[1];
            _Float16* d = &bS[brow][bq * 8];
#pragma unroll
            for (int i = 0; i < 4; ++i) {
                d[i]     = (_Float16)f0[i];
                d[4 + i] = (_Float16)f1[i];
            }
        }
        __syncthreads();

        // ---- A fragment (ISA 7.12.2: lane holds row L&15; K runs
        //      [half*8, +8) and [16+half*8, +8)) ----
        v8h alo = *(const v8h*)&aS[wave * 16 + r16][half * 8];
        v8h ahi = *(const v8h*)&aS[wave * 16 + r16][16 + half * 8];
        v16h a  = SHUF16(alo, ahi);

#pragma unroll
        for (int tt = 0; tt < 4; ++tt) {
            v8h blo = *(const v8h*)&bS[tt * 16 + r16][half * 8];
            v8h bhi = *(const v8h*)&bS[tt * 16 + r16][16 + half * 8];
            v16h b  = SHUF16(blo, bhi);
            acc[tt] = __builtin_amdgcn_wmma_f32_16x16x32_f16(false, a, false, b,
                                                             (short)0, acc[tt], false, false);
        }
    }

    // ---- epilogue: 4 tiles of 16x16 ----
#pragma unroll
    for (int tt = 0; tt < 4; ++tt) {
        const int col  = n0b + tt * 16 + r16;
        const float bv = bias ? bias[col] : 0.f;
#pragma unroll
        for (int r = 0; r < 8; ++r) {
            const int m = m0b + wave * 16 + (half << 3) + r;
            float v = acc[tt][r] + bv;
            if (act == 1) v = v / (1.f + expf(-v));               // silu
            if (addend)   v += addend[(size_t)m * ldadd + col];   // residual
            if (rowscale) v *= rowscale[(size_t)m * rsStride];    // per-token gate
            const size_t idx = (size_t)m * ldc + col;
            if (accumulate) C[idx] += v; else C[idx] = v;
        }
    }
}

// =====================================================================
// WMMA GEMM, NN:  C[M,N] = A[M,K] @ B[K,N]   (for P @ V only, ~3% of flops)
// =====================================================================
__global__ void gemm_nn(const float* __restrict__ A, int lda,
                        const float* __restrict__ B, int ldb,
                        float* __restrict__ C, int ldc,
                        int M, int N, int K)
{
    const int lane = threadIdx.x & 31;
    const int wave = threadIdx.x >> 5;
    const int m0 = blockIdx.y * 128 + wave * 16;
    const int n0 = blockIdx.x * 16;
    const int half = lane >> 4;
    const int r16  = lane & 15;

    const float* Arow = A + (size_t)(m0 + r16) * lda;
    v8f acc = {};
    for (int k0 = 0; k0 < K; k0 += 32) {
        const float* pa0 = Arow + k0 + half * 8;
        const float* pb0 = B + (size_t)(k0 + half * 8) * ldb + (n0 + r16);
        __builtin_prefetch(pa0 + 32, 0, 3);
        v16h a, b;
#pragma unroll
        for (int i = 0; i < 8; ++i) {
            a[i]     = (_Float16)pa0[i];
            a[8 + i] = (_Float16)pa0[16 + i];
            b[i]     = (_Float16)pb0[(size_t)i * ldb];
            b[8 + i] = (_Float16)pb0[(size_t)(16 + i) * ldb];
        }
        acc = __builtin_amdgcn_wmma_f32_16x16x32_f16(false, a, false, b,
                                                     (short)0, acc, false, false);
    }
    const int col = n0 + r16;
#pragma unroll
    for (int r = 0; r < 8; ++r) {
        const int m = m0 + (half << 3) + r;
        C[(size_t)m * ldc + col] = acc[r];
    }
}

// =====================================================================
// RMSNorm: one block per row; safe in-place.
// =====================================================================
__global__ void rmsnorm_kernel(const float* __restrict__ in,
                               const float* __restrict__ w,
                               float* __restrict__ out, int D)
{
    const int row = blockIdx.x;
    const float* p = in + (size_t)row * D;
    float ss = 0.f;
    for (int i = threadIdx.x; i < D; i += blockDim.x) { float v = p[i]; ss += v * v; }
    __shared__ float red[256];
    red[threadIdx.x] = ss; __syncthreads();
    for (int o = 128; o > 0; o >>= 1) {
        if (threadIdx.x < o) red[threadIdx.x] += red[threadIdx.x + o];
        __syncthreads();
    }
    const float inv = rsqrtf(red[0] / (float)D + 1e-6f);
    for (int i = threadIdx.x; i < D; i += blockDim.x)
        out[(size_t)row * D + i] = p[i] * inv * w[i];
}

// =====================================================================
// RoPE (interleaved pairs). x layout: [S, H, DR], halfD = DR/2 = 32
// =====================================================================
__global__ void rope_kernel(float* __restrict__ x,
                            const float* __restrict__ cosb,
                            const float* __restrict__ sinb,
                            int H, int halfD, int total)
{
    const int idx = blockIdx.x * blockDim.x + threadIdx.x;
    if (idx >= total) return;
    const int i  = idx % halfD;
    const int sh = idx / halfD;
    const int s  = sh / H;
    const float c  = cosb[s * halfD + i];
    const float sn = sinb[s * halfD + i];
    float* p = x + (size_t)sh * (2 * halfD) + 2 * i;
    const float x0 = p[0], x1 = p[1];
    p[0] = x0 * c - x1 * sn;
    p[1] = x0 * sn + x1 * c;
}

// =====================================================================
// Build q=[qC,qR], k=[kR(broadcast),kC] exactly as the reference concatenates
// =====================================================================
__global__ void build_qk(const float* __restrict__ qC, const float* __restrict__ qR,
                         const float* __restrict__ kr, const float* __restrict__ kC,
                         float* __restrict__ q, float* __restrict__ k)
{
    const int idx = blockIdx.x * blockDim.x + threadIdx.x;
    const int total = SQ * NHD * QKD;
    if (idx >= total) return;
    const int j  = idx % QKD;
    const int sh = idx / QKD;
    const int h  = sh % NHD;
    const int s  = sh / NHD;
    q[idx] = (j < DH_) ? qC[(size_t)s * (NHD * DH_) + h * DH_ + j]
                       : qR[(size_t)(s * NHD + h) * DR_ + (j - DH_)];
    k[idx] = (j < DR_) ? kr[(size_t)s * DR_ + j]
                       : kC[(size_t)s * (NHD * DH_) + h * DH_ + (j - DR_)];
}

// =====================================================================
// Causal softmax over one score row (with scale); masked tail -> 0
// =====================================================================
__global__ void softmax_causal(float* __restrict__ sc, int n, float scale)
{
    const int row = blockIdx.x;
    float* p = sc + (size_t)row * n;
    const int valid = row + 1;
    __shared__ float red[256];

    float mx = -3.4e38f;
    for (int i = threadIdx.x; i < valid; i += blockDim.x)
        mx = fmaxf(mx, p[i] * scale);
    red[threadIdx.x] = mx; __syncthreads();
    for (int o = 128; o > 0; o >>= 1) {
        if (threadIdx.x < o) red[threadIdx.x] = fmaxf(red[threadIdx.x], red[threadIdx.x + o]);
        __syncthreads();
    }
    mx = red[0]; __syncthreads();

    float ssum = 0.f;
    for (int i = threadIdx.x; i < valid; i += blockDim.x) {
        const float e = expf(p[i] * scale - mx);
        p[i] = e; ssum += e;
    }
    red[threadIdx.x] = ssum; __syncthreads();
    for (int o = 128; o > 0; o >>= 1) {
        if (threadIdx.x < o) red[threadIdx.x] += red[threadIdx.x + o];
        __syncthreads();
    }
    const float inv = 1.f / red[0];
    for (int i = threadIdx.x; i < valid; i += blockDim.x) p[i] *= inv;
    for (int i = valid + threadIdx.x; i < n; i += blockDim.x) p[i] = 0.f;
}

// =====================================================================
// Router: sigmoid affinities, top-2 on (affinity+bias), softmax gates on
// the raw affinities (matching the reference), scatter into combine[S,E]
// =====================================================================
__global__ void route_kernel(const float* __restrict__ y,
                             const float* __restrict__ centroids,
                             const float* __restrict__ biases,
                             float* __restrict__ combine)
{
    const int s = blockIdx.x;
    __shared__ float red[256];
    __shared__ float aff[NE];
    for (int e = 0; e < NE; ++e) {
        float pr = 0.f;
        for (int i = threadIdx.x; i < DM; i += blockDim.x)
            pr += y[(size_t)s * DM + i] * centroids[(size_t)e * DM + i];
        red[threadIdx.x] = pr; __syncthreads();
        for (int o = 128; o > 0; o >>= 1) {
            if (threadIdx.x < o) red[threadIdx.x] += red[threadIdx.x + o];
            __syncthreads();
        }
        if (threadIdx.x == 0) aff[e] = 1.f / (1.f + expf(-red[0]));
        __syncthreads();
    }
    if (threadIdx.x == 0) {
        int   i0 = 0, i1 = -1;
        float b0 = -3.4e38f, b1 = -3.4e38f;
        for (int e = 0; e < NE; ++e) {
            const float v = aff[e] + biases[e];
            if (v > b0) { b1 = b0; i1 = i0; b0 = v; i0 = e; }
            else if (v > b1) { b1 = v; i1 = e; }
        }
        const float a0 = aff[i0], a1 = aff[i1];
        const float mx = fmaxf(a0, a1);
        const float e0 = expf(a0 - mx), e1 = expf(a1 - mx);
        const float inv = 1.f / (e0 + e1);
        for (int e = 0; e < NE; ++e) combine[(size_t)s * NE + e] = 0.f;
        combine[(size_t)s * NE + i0] = e0 * inv;
        combine[(size_t)s * NE + i1] = e1 * inv;
    }
}

// ---------------- elementwise glue ----------------
__global__ void mul_inplace(float* __restrict__ a, const float* __restrict__ b, int n)
{
    const int i = blockIdx.x * blockDim.x + threadIdx.x;
    if (i < n) a[i] *= b[i];
}

__global__ void final_add(const float* __restrict__ a, const float* __restrict__ b,
                          const float* __restrict__ c, float* __restrict__ out, int n)
{
    const int i = blockIdx.x * blockDim.x + threadIdx.x;
    if (i < n) out[i] = a[i] + b[i] + c[i];
}

// =====================================================================
// Host-side orchestration
// =====================================================================
extern "C" void kernel_launch(void* const* d_in, const int* in_sizes, int n_in,
                              void* d_out, int out_size, void* d_ws, size_t ws_size,
                              hipStream_t stream)
{
    (void)in_sizes; (void)n_in; (void)out_size; (void)ws_size;

    const float* h_in     = (const float*)d_in[0];
    const float* fcos     = (const float*)d_in[1];
    const float* fsin     = (const float*)d_in[2];
    /* d_in[3] = mask (bool): causal, computed analytically */
    const float* attn_nw  = (const float*)d_in[4];
    const float* ffn_nw   = (const float*)d_in[5];
    const float* wq_down  = (const float*)d_in[6];
    const float* bq_down  = (const float*)d_in[7];
    const float* q_norm_w = (const float*)d_in[8];
    const float* wqc      = (const float*)d_in[9];
    const float* bqc      = (const float*)d_in[10];
    const float* wqr      = (const float*)d_in[11];
    const float* bqr      = (const float*)d_in[12];
    const float* wkr      = (const float*)d_in[13];
    const float* bkr      = (const float*)d_in[14];
    const float* kr_nw    = (const float*)d_in[15];
    const float* wkv_down = (const float*)d_in[16];
    const float* bkv_down = (const float*)d_in[17];
    const float* kv_nw    = (const float*)d_in[18];
    const float* wkc      = (const float*)d_in[19];
    const float* bkc      = (const float*)d_in[20];
    const float* wvc      = (const float*)d_in[21];
    const float* bvc      = (const float*)d_in[22];
    const float* wo       = (const float*)d_in[23];
    const float* bo       = (const float*)d_in[24];
    const float* sw1      = (const float*)d_in[25];
    const float* sb1      = (const float*)d_in[26];
    const float* sw2      = (const float*)d_in[27];
    const float* sb2      = (const float*)d_in[28];
    const float* sw3      = (const float*)d_in[29];
    const float* sb3      = (const float*)d_in[30];
    const float* ew1      = (const float*)d_in[31];
    const float* eb1      = (const float*)d_in[32];
    const float* ew2      = (const float*)d_in[33];
    const float* eb2      = (const float*)d_in[34];
    const float* ew3      = (const float*)d_in[35];
    const float* eb3      = (const float*)d_in[36];
    const float* cent     = (const float*)d_in[37];
    const float* biases   = (const float*)d_in[38];

    float* out = (float*)d_out;

    // ---- workspace layout (~65 MB fp32) ----
    float* Wp = (float*)d_ws;
    size_t off = 0;
    auto alloc = [&](size_t n) { float* p = Wp + off; off += n; return p; };
    float* x      = alloc((size_t)SQ * DM);
    float* cQ     = alloc((size_t)SQ * DCQ_);
    float* qR     = alloc((size_t)SQ * (NHD * DR_));
    float* qC     = alloc((size_t)SQ * (NHD * DH_));
    float* kr     = alloc((size_t)SQ * DR_);
    float* ckv    = alloc((size_t)SQ * DC_);
    float* kC     = alloc((size_t)SQ * (NHD * DH_));
    float* vC     = alloc((size_t)SQ * (NHD * DH_));
    float* qbuf   = alloc((size_t)SQ * NHD * QKD);
    float* kbuf   = alloc((size_t)SQ * NHD * QKD);
    float* scores = alloc((size_t)SQ * SQ);          // reused per head
    float* attn   = alloc((size_t)SQ * (NHD * DH_));
    float* x1     = alloc((size_t)SQ * DM);
    float* y      = alloc((size_t)SQ * DM);
    float* t1     = alloc((size_t)SQ * INTER_);
    float* t3     = alloc((size_t)SQ * INTER_);
    float* shrd   = alloc((size_t)SQ * DM);
    float* comb   = alloc((size_t)SQ * NE);
    float* routed = alloc((size_t)SQ * DM);

    const dim3 blk(256);
    auto ggrid = [](int N) { return dim3(N / 64, SQ / 128); };   // 128x64 block tiles
    const int M = SQ;

    // 1) x = rmsnorm(h, attn_norm_w)
    rmsnorm_kernel<<<SQ, blk, 0, stream>>>(h_in, attn_nw, x, DM);

    // 2) cQ = rmsnorm(x @ wq_down^T + b, q_norm_w)
    gemm_nt<<<ggrid(DCQ_), blk, 0, stream>>>(x, DM, wq_down, DM, bq_down, cQ, DCQ_,
                                             M, DCQ_, DM, 0, nullptr, 0, nullptr, 0, 0);
    rmsnorm_kernel<<<SQ, blk, 0, stream>>>(cQ, q_norm_w, cQ, DCQ_);

    // 3) qR = rope(cQ @ wqr^T + b); qC = cQ @ wqc^T + b
    gemm_nt<<<ggrid(NHD * DR_), blk, 0, stream>>>(cQ, DCQ_, wqr, DCQ_, bqr, qR, NHD * DR_,
                                                  M, NHD * DR_, DCQ_, 0, nullptr, 0, nullptr, 0, 0);
    gemm_nt<<<ggrid(NHD * DH_), blk, 0, stream>>>(cQ, DCQ_, wqc, DCQ_, bqc, qC, NHD * DH_,
                                                  M, NHD * DH_, DCQ_, 0, nullptr, 0, nullptr, 0, 0);
    {
        int tot = SQ * NHD * (DR_ / 2);
        rope_kernel<<<(tot + 255) / 256, blk, 0, stream>>>(qR, fcos, fsin, NHD, DR_ / 2, tot);
    }

    // 4) kR = rope(rmsnorm(x @ wkr^T + b, kr_norm_w))
    gemm_nt<<<ggrid(DR_), blk, 0, stream>>>(x, DM, wkr, DM, bkr, kr, DR_,
                                            M, DR_, DM, 0, nullptr, 0, nullptr, 0, 0);
    rmsnorm_kernel<<<SQ, blk, 0, stream>>>(kr, kr_nw, kr, DR_);
    {
        int tot = SQ * 1 * (DR_ / 2);
        rope_kernel<<<(tot + 255) / 256, blk, 0, stream>>>(kr, fcos, fsin, 1, DR_ / 2, tot);
    }

    // 5) cKV = rmsnorm(x @ wkv_down^T + b); kC, vC
    gemm_nt<<<ggrid(DC_), blk, 0, stream>>>(x, DM, wkv_down, DM, bkv_down, ckv, DC_,
                                            M, DC_, DM, 0, nullptr, 0, nullptr, 0, 0);
    rmsnorm_kernel<<<SQ, blk, 0, stream>>>(ckv, kv_nw, ckv, DC_);
    gemm_nt<<<ggrid(NHD * DH_), blk, 0, stream>>>(ckv, DC_, wkc, DC_, bkc, kC, NHD * DH_,
                                                  M, NHD * DH_, DC_, 0, nullptr, 0, nullptr, 0, 0);
    gemm_nt<<<ggrid(NHD * DH_), blk, 0, stream>>>(ckv, DC_, wvc, DC_, bvc, vC, NHD * DH_,
                                                  M, NHD * DH_, DC_, 0, nullptr, 0, nullptr, 0, 0);

    // 6) q = [qC,qR], k = [kR,kC]
    {
        int tot = SQ * NHD * QKD;
        build_qk<<<(tot + 255) / 256, blk, 0, stream>>>(qC, qR, kr, kC, qbuf, kbuf);
    }

    // 7) attention, one head at a time (reuse score buffer)
    const float scale = 1.f / sqrtf((float)QKD);
    for (int h = 0; h < NHD; ++h) {
        gemm_nt<<<ggrid(SQ), blk, 0, stream>>>(qbuf + h * QKD, NHD * QKD,
                                               kbuf + h * QKD, NHD * QKD, nullptr,
                                               scores, SQ, M, SQ, QKD,
                                               0, nullptr, 0, nullptr, 0, 0);
        softmax_causal<<<SQ, blk, 0, stream>>>(scores, SQ, scale);
        gemm_nn<<<dim3(DH_ / 16, SQ / 128), blk, 0, stream>>>(scores, SQ,
                                                              vC + h * DH_, NHD * DH_,
                                                              attn + h * DH_, NHD * DH_,
                                                              M, DH_, SQ);
    }

    // 8) x1 = h + attn @ wo^T + bo
    gemm_nt<<<ggrid(DM), blk, 0, stream>>>(attn, NHD * DH_, wo, NHD * DH_, bo, x1, DM,
                                           M, DM, NHD * DH_, 0, h_in, DM, nullptr, 0, 0);

    // 9) y = rmsnorm(x1, ffn_norm_w)
    rmsnorm_kernel<<<SQ, blk, 0, stream>>>(x1, ffn_nw, y, DM);

    // 10) shared FFN: silu(y@sw1+b) * (y@sw3+b) @ sw2 + b
    gemm_nt<<<ggrid(INTER_), blk, 0, stream>>>(y, DM, sw1, DM, sb1, t1, INTER_,
                                               M, INTER_, DM, 1, nullptr, 0, nullptr, 0, 0);
    gemm_nt<<<ggrid(INTER_), blk, 0, stream>>>(y, DM, sw3, DM, sb3, t3, INTER_,
                                               M, INTER_, DM, 0, nullptr, 0, nullptr, 0, 0);
    mul_inplace<<<(SQ * INTER_ + 255) / 256, blk, 0, stream>>>(t1, t3, SQ * INTER_);
    gemm_nt<<<ggrid(DM), blk, 0, stream>>>(t1, INTER_, sw2, INTER_, sb2, shrd, DM,
                                           M, DM, INTER_, 0, nullptr, 0, nullptr, 0, 0);

    // 11) routing gates
    route_kernel<<<SQ, blk, 0, stream>>>(y, cent, biases, comb);

    // 12) MoE: dense over all experts, gated accumulate into routed
    for (int e = 0; e < NE; ++e) {
        const float* e1 = ew1 + (size_t)e * INTER_ * DM;
        const float* e3 = ew3 + (size_t)e * INTER_ * DM;
        const float* e2 = ew2 + (size_t)e * DM * INTER_;
        gemm_nt<<<ggrid(INTER_), blk, 0, stream>>>(y, DM, e1, DM, eb1 + e * INTER_, t1, INTER_,
                                                   M, INTER_, DM, 1, nullptr, 0, nullptr, 0, 0);
        gemm_nt<<<ggrid(INTER_), blk, 0, stream>>>(y, DM, e3, DM, eb3 + e * INTER_, t3, INTER_,
                                                   M, INTER_, DM, 0, nullptr, 0, nullptr, 0, 0);
        mul_inplace<<<(SQ * INTER_ + 255) / 256, blk, 0, stream>>>(t1, t3, SQ * INTER_);
        gemm_nt<<<ggrid(DM), blk, 0, stream>>>(t1, INTER_, e2, INTER_, eb2 + e * DM, routed, DM,
                                               M, DM, INTER_, 0, nullptr, 0,
                                               comb + e, NE, /*accumulate=*/(e > 0) ? 1 : 0);
    }

    // 13) out = x1 + shared + routed
    final_add<<<(SQ * DM + 255) / 256, blk, 0, stream>>>(x1, shrd, routed, out, SQ * DM);
}
